// GraphAttentionLayer_63952063037634
// MI455X (gfx1250) — compile-verified, matrix operand order checked by reading, not run on hardware
//
#include <hip/hip_runtime.h>
#include <hip/hip_bf16.h>

typedef __attribute__((ext_vector_type(16))) _Float16 v16h;
typedef __attribute__((ext_vector_type(2)))  __fp16   fp16x2;
typedef __attribute__((ext_vector_type(8)))  float    v8f;

#define NB    2
#define NN    2048
#define NH    8
#define ND    64
#define FIN   512
#define MODEL 512      // NH*ND
#define SCALE 0.125f   // D^-0.5

union F16Frag { v16h v; unsigned u[8]; };

// pack two f32 -> one dword of two f16 via v_cvt_pk_f16_f32
__device__ __forceinline__ unsigned pk2(float a, float b) {
  fp16x2 r = __builtin_amdgcn_cvt_pkrtz(a, b);
  unsigned u;
  __builtin_memcpy(&u, &r, 4);
  return u;
}

// ---------------------------------------------------------------------------
// Kernel 1: h = x @ W + b   (f32 in, f16 out, stored [B,H,N,D])
// One wave computes one 16x16 output tile via v_wmma_f32_16x16x32_f16.
// ---------------------------------------------------------------------------
__global__ void proj_kernel(const float* __restrict__ x, const float* __restrict__ W,
                            const float* __restrict__ bias, _Float16* __restrict__ ht) {
  const int lane    = threadIdx.x & 31;
  const int wv      = threadIdx.x >> 5;
  const int tile    = blockIdx.x * 8 + wv;       // 8192 tiles total
  const int mt      = tile >> 5;                 // 0..255  (rows of 16)
  const int nt      = tile & 31;                 // 0..31   (cols of 16)
  const int halfsel = (lane & 16) ? 8 : 0;
  const int l15     = lane & 15;
  const int rowA    = mt * 16 + l15;             // A-matrix: lane = row
  const int colB    = nt * 16 + l15;             // B-matrix: lane = col

  v8f acc = {};
  #pragma unroll 4
  for (int kk = 0; kk < FIN; kk += 32) {
    F16Frag a, b;
    #pragma unroll
    for (int i = 0; i < 8; ++i) {
      const int e = 2 * i;
      const int k = kk + (e < 8 ? e : e + 8) + halfsel;     // ISA 16-bit A layout
      const float2 f2 = *reinterpret_cast<const float2*>(x + (size_t)rowA * FIN + k);
      a.u[i] = pk2(f2.x, f2.y);                             // v_cvt_pk_f16_f32
      const float w0 = W[(size_t)k * MODEL + colB];
      const float w1 = W[(size_t)(k + 1) * MODEL + colB];
      b.u[i] = pk2(w0, w1);
    }
    acc = __builtin_amdgcn_wmma_f32_16x16x32_f16(false, a.v, false, b.v,
                                                 (short)0, acc, false, false);
  }

  // C/D layout: lanes 0-15 -> M=r, lanes 16-31 -> M=r+8; N = lane&15
  #pragma unroll
  for (int r = 0; r < 8; ++r) {
    const int m    = r + halfsel;
    const int Rrow = mt * 16 + m;
    const int col  = nt * 16 + l15;
    const float v  = acc[r] + bias[col];
    const int bb = Rrow >> 11, nr = Rrow & (NN - 1);
    const int hh = col >> 6,  dd = col & 63;
    ht[(((size_t)(bb * NH + hh) * NN) + nr) * ND + dd] = (_Float16)v;
  }
}

// ---------------------------------------------------------------------------
// Kernel 2: fused masked attention. One workgroup per (b, 16-row tile);
// wave w owns head w. Two passes: (1) softmax denominators, (2) P, attn-mean
// (fixed-order cross-wave reduce => deterministic), and O = P @ V.
// ---------------------------------------------------------------------------
__global__ void attn_kernel(const _Float16* __restrict__ ht, const int* __restrict__ adj,
                            float* __restrict__ oacc, float* __restrict__ attn_mean) {
  __shared__ __align__(16) _Float16 Pbuf[8 * 512];  // per-wave 16x32 P tile (f16, A-layout src)
  __shared__ int adjs[512];                         // shared 16x32 adjacency tile

  const int lane    = threadIdx.x & 31;
  const int w       = threadIdx.x >> 5;      // head
  const int t       = threadIdx.x;
  const int bb      = blockIdx.x >> 7;
  const int mt      = blockIdx.x & 127;
  const int rowbase = mt * 16;
  const int halfsel = (lane & 16) ? 8 : 0;
  const int l15     = lane & 15;

  const _Float16* hb   = ht + (size_t)(bb * NH + w) * NN * ND;           // [N,D] head slice
  const int*      adjb = adj + (size_t)bb * NN * NN + (size_t)rowbase * NN;
  float*          mnb  = attn_mean + (size_t)bb * NN * NN + (size_t)rowbase * NN;

  // Q fragments (A 16x32, two halves of D=64), loaded once
  F16Frag qf0, qf1;
  {
    const _Float16* qrow = hb + (size_t)(rowbase + l15) * ND;
    #pragma unroll
    for (int i = 0; i < 8; ++i) {
      const int k = (2 * i < 8 ? 2 * i : 2 * i + 8) + halfsel;
      qf0.u[i] = *reinterpret_cast<const unsigned*>(qrow + k);
      qf1.u[i] = *reinterpret_cast<const unsigned*>(qrow + 32 + k);
    }
  }

  const v8f zacc = {};

  // ---- Pass 1: un-normalized softmax row sums (masked -> 0 contribution) ----
  float rs[8];
  #pragma unroll
  for (int r = 0; r < 8; ++r) rs[r] = 0.f;

  for (int jp = 0; jp < NN / 32; ++jp) {
    const int colbase = jp * 32;
    __syncthreads();
    adjs[t]       = adjb[(size_t)(t >> 5) * NN + colbase + (t & 31)];
    adjs[t + 256] = adjb[(size_t)((t + 256) >> 5) * NN + colbase + ((t + 256) & 31)];
    if (jp + 1 < NN / 32)
      __builtin_prefetch(adjb + (size_t)(t >> 5) * NN + colbase + 32 + (t & 31), 0, 0);
    __syncthreads();
    #pragma unroll
    for (int sub = 0; sub < 2; ++sub) {
      const int cb = colbase + sub * 16;
      const _Float16* krow = hb + (size_t)(cb + l15) * ND;   // B: lane = K-row (col of S)
      F16Frag kf0, kf1;
      #pragma unroll
      for (int i = 0; i < 8; ++i) {
        const int k = (2 * i < 8 ? 2 * i : 2 * i + 8) + halfsel;
        kf0.u[i] = *reinterpret_cast<const unsigned*>(krow + k);
        kf1.u[i] = *reinterpret_cast<const unsigned*>(krow + 32 + k);
      }
      // independent accumulators: no WMMA->WMMA RAW hazard NOPs
      v8f s0 = __builtin_amdgcn_wmma_f32_16x16x32_f16(false, qf0.v, false, kf0.v, (short)0, zacc, false, false);
      v8f s1 = __builtin_amdgcn_wmma_f32_16x16x32_f16(false, qf1.v, false, kf1.v, (short)0, zacc, false, false);
      #pragma unroll
      for (int r = 0; r < 8; ++r) {
        const int m  = r + halfsel;
        const int av = adjs[m * 32 + sub * 16 + l15];
        rs[r] += av ? __expf((s0[r] + s1[r]) * SCALE) : 0.f;
      }
    }
  }
  #pragma unroll
  for (int r = 0; r < 8; ++r) {     // reduce across the 16-lane half owning the row
    float v = rs[r];
    v += __shfl_xor(v, 1, 16);
    v += __shfl_xor(v, 2, 16);
    v += __shfl_xor(v, 4, 16);
    v += __shfl_xor(v, 8, 16);
    rs[r] = 1.0f / v;               // 1/denominator
  }

  // ---- Pass 2: P, attn-mean, O = P @ V ----
  v8f o0 = {}, o1 = {}, o2 = {}, o3 = {};
  for (int jp = 0; jp < NN / 32; ++jp) {
    const int colbase = jp * 32;
    __syncthreads();
    adjs[t]       = adjb[(size_t)(t >> 5) * NN + colbase + (t & 31)];
    adjs[t + 256] = adjb[(size_t)((t + 256) >> 5) * NN + colbase + ((t + 256) & 31)];
    if (jp + 1 < NN / 32)           // global_prefetch_b8 for next adj tile
      __builtin_prefetch(adjb + (size_t)(t >> 5) * NN + colbase + 32 + (t & 31), 0, 0);
    __syncthreads();

    #pragma unroll
    for (int sub = 0; sub < 2; ++sub) {
      const int cb = colbase + sub * 16;
      const _Float16* krow = hb + (size_t)(cb + l15) * ND;
      F16Frag kf0, kf1;
      #pragma unroll
      for (int i = 0; i < 8; ++i) {
        const int k = (2 * i < 8 ? 2 * i : 2 * i + 8) + halfsel;
        kf0.u[i] = *reinterpret_cast<const unsigned*>(krow + k);
        kf1.u[i] = *reinterpret_cast<const unsigned*>(krow + 32 + k);
      }
      v8f s0 = __builtin_amdgcn_wmma_f32_16x16x32_f16(false, qf0.v, false, kf0.v, (short)0, zacc, false, false);
      v8f s1 = __builtin_amdgcn_wmma_f32_16x16x32_f16(false, qf1.v, false, kf1.v, (short)0, zacc, false, false);
      #pragma unroll
      for (int r = 0; r < 8; ++r) {
        const int m  = r + halfsel;
        const int av = adjs[m * 32 + sub * 16 + l15];
        const float p = av ? __expf((s0[r] + s1[r]) * SCALE) * rs[r] : 0.f;
        Pbuf[w * 512 + m * 32 + sub * 16 + l15] = (_Float16)p;
      }
    }
    __syncthreads();

    // attn.mean over heads: fixed-order sum -> deterministic, no atomics
    #pragma unroll
    for (int half = 0; half < 2; ++half) {
      const int lin = t + half * 256;
      float msum = 0.f;
      #pragma unroll
      for (int ww = 0; ww < 8; ++ww) msum += (float)Pbuf[ww * 512 + lin];
      mnb[(size_t)(lin >> 5) * NN + colbase + (lin & 31)] = msum * 0.125f;
    }

    // A-fragment rebuild from f16 LDS: pairs (k,k+1) contiguous -> 8x ds_load_b32
    F16Frag af;
    {
      const unsigned* pw = reinterpret_cast<const unsigned*>(Pbuf + w * 512 + l15 * 32);
      #pragma unroll
      for (int i = 0; i < 8; ++i) {
        const int k = (2 * i < 8 ? 2 * i : 2 * i + 8) + halfsel;
        af.u[i] = pw[k >> 1];
      }
    }
    // V fragments (B 32x16 per 16-dim slice) and accumulate
    #pragma unroll
    for (int dt = 0; dt < 4; ++dt) {
      v16h vf;
      #pragma unroll
      for (int e = 0; e < 16; ++e) {
        const int k = (e < 8 ? e : e + 8) + halfsel;
        vf[e] = hb[(size_t)(colbase + k) * ND + dt * 16 + l15];
      }
      v8f* op = (dt == 0) ? &o0 : (dt == 1) ? &o1 : (dt == 2) ? &o2 : &o3;
      *op = __builtin_amdgcn_wmma_f32_16x16x32_f16(false, af.v, false, vf, (short)0, *op, false, false);
    }
  }

  // store O (pre-LayerNorm), f32, [B*N, MODEL]
  float* ob = oacc + (size_t)(bb * NN + rowbase) * MODEL + w * ND;
  #pragma unroll
  for (int r = 0; r < 8; ++r) {
    const int m = r + halfsel;
    ob[(size_t)m * MODEL +  0 + l15] = o0[r];
    ob[(size_t)m * MODEL + 16 + l15] = o1[r];
    ob[(size_t)m * MODEL + 32 + l15] = o2[r];
    ob[(size_t)m * MODEL + 48 + l15] = o3[r];
  }
}

// ---------------------------------------------------------------------------
// Kernel 3: LayerNorm over MODEL=512, one block per row
// ---------------------------------------------------------------------------
__global__ void ln_kernel(const float* __restrict__ in, const float* __restrict__ gamma,
                          const float* __restrict__ beta, float* __restrict__ out) {
  __shared__ float red[256];
  const int t = threadIdx.x;
  const size_t base = (size_t)blockIdx.x * MODEL;
  const float v0 = in[base + t];
  const float v1 = in[base + 256 + t];

  red[t] = v0 + v1;
  __syncthreads();
  for (int sft = 128; sft > 0; sft >>= 1) {
    if (t < sft) red[t] += red[t + sft];
    __syncthreads();
  }
  const float mu = red[0] * (1.0f / MODEL);
  __syncthreads();
  red[t] = (v0 - mu) * (v0 - mu) + (v1 - mu) * (v1 - mu);
  __syncthreads();
  for (int sft = 128; sft > 0; sft >>= 1) {
    if (t < sft) red[t] += red[t + sft];
    __syncthreads();
  }
  const float rstd = rsqrtf(red[0] * (1.0f / MODEL) + 1e-5f);
  out[base + t]       = (v0 - mu) * rstd * gamma[t]       + beta[t];
  out[base + 256 + t] = (v1 - mu) * rstd * gamma[t + 256] + beta[t + 256];
}

// ---------------------------------------------------------------------------
extern "C" void kernel_launch(void* const* d_in, const int* in_sizes, int n_in,
                              void* d_out, int out_size, void* d_ws, size_t ws_size,
                              hipStream_t stream) {
  (void)in_sizes; (void)n_in; (void)out_size; (void)ws_size;
  const float* x     = (const float*)d_in[0];
  const int*   adj   = (const int*)d_in[1];
  const float* W     = (const float*)d_in[2];
  const float* Wb    = (const float*)d_in[3];
  const float* gamma = (const float*)d_in[4];
  const float* beta  = (const float*)d_in[5];

  float* out0      = (float*)d_out;                               // [B,N,MODEL]
  float* attn_mean = out0 + (size_t)NB * NN * MODEL;              // [B,N,N]

  _Float16* ht = (_Float16*)d_ws;                                 // 4 MB  [B,H,N,D] f16
  float* oacc  = (float*)((char*)d_ws +
                          (size_t)NB * NN * MODEL * sizeof(_Float16));  // 8 MB pre-LN out

  proj_kernel<<<1024, 256, 0, stream>>>(x, W, Wb, ht);
  attn_kernel<<<NB * (NN / 16), 256, 0, stream>>>(ht, adj, oacc, attn_mean);
  ln_kernel<<<NB * NN, 256, 0, stream>>>(oacc, gamma, beta, out0);
}
